// MyNet_7284264534507
// MI455X (gfx1250) — matrix operand order repaired and checked by reading, not compiled, not run on hardware
//
#include <hip/hip_runtime.h>
#include <hip/hip_bf16.h>

// ---------------------------------------------------------------------------
// MyNet on MI455X (gfx1250, wave32, WMMA).
// Dominant cost: 8 MLPs (3->64->128->256->1024) over 32768 points (~160 GFLOP)
// -> f16 WMMA GEMMs with f32 accumulation. Last (256->1024) layer fused with
// the low-rank contractions to avoid 64MB intermediates.
// ---------------------------------------------------------------------------

typedef __attribute__((ext_vector_type(16))) _Float16 v16h;
typedef __attribute__((ext_vector_type(8)))  _Float16 v8h;
typedef __attribute__((ext_vector_type(8)))  float    v8f;

union V16 { v16h v; v8h h[2]; };

#define MTOT   32768      // B*N
#define NPTS   4096       // points per batch element
#define WIDTH  32
#define BN_EPS 1e-5f

// ===========================================================================
// fc0: h = v @ W(3x32) + b   (no relu), outputs f32 + f16
// ===========================================================================
__global__ __launch_bounds__(256) void k_fc0(
    const float* __restrict__ v, const float* __restrict__ W,
    const float* __restrict__ b, float* __restrict__ h32,
    _Float16* __restrict__ h16)
{
    __shared__ float Ws[96], bs[32];
    const int tid = threadIdx.x;
    if (tid < 96) Ws[tid] = W[tid];
    if (tid < 32) bs[tid] = b[tid];
    __syncthreads();
    const int p = blockIdx.x * 256 + tid;
    const float x0 = v[(size_t)p*3+0], x1 = v[(size_t)p*3+1], x2 = v[(size_t)p*3+2];
    #pragma unroll
    for (int o = 0; o < 32; ++o) {
        const float y = bs[o] + x0*Ws[o] + x1*Ws[32+o] + x2*Ws[64+o];
        h32[(size_t)p*32 + o] = y;
        h16[(size_t)p*32 + o] = (_Float16)y;
    }
}

// ===========================================================================
// MLP first layer: act1 = relu(a @ W(3x64) + b) as f16
// ===========================================================================
__global__ __launch_bounds__(256) void k_mlp_first(
    const float* __restrict__ v, const float* __restrict__ W,
    const float* __restrict__ b, _Float16* __restrict__ Y)
{
    __shared__ float Ws[192], bs[64];
    const int tid = threadIdx.x;
    if (tid < 192) Ws[tid] = W[tid];
    if (tid < 64)  bs[tid] = b[tid];
    __syncthreads();
    const int p = blockIdx.x * 256 + tid;
    const float x0 = v[(size_t)p*3+0], x1 = v[(size_t)p*3+1], x2 = v[(size_t)p*3+2];
    #pragma unroll
    for (int o = 0; o < 64; ++o) {
        float y = bs[o] + x0*Ws[o] + x1*Ws[64+o] + x2*Ws[128+o];
        y = fmaxf(y, 0.f);
        Y[(size_t)p*64 + o] = (_Float16)y;
    }
}

// ===========================================================================
// Generic WMMA GEMM: Y(f16) = relu(X(f16, MxK) @ W(f32, KxN) + bias)
// 256 threads = 8 waves; tile 128x64; wave = 32x32 (2x2 wmma frags); Kstep 32.
// ===========================================================================
__global__ __launch_bounds__(256) void k_gemm_wmma(
    const _Float16* __restrict__ X, const float* __restrict__ W,
    const float* __restrict__ bias, _Float16* __restrict__ Y,
    int K, int Nout, int relu)
{
    __shared__ __align__(16) _Float16 As[128*40];   // A tile, rows padded to 40
    __shared__ __align__(16) _Float16 Bs[64*40];    // B tile transposed [n][k]
    const int tid  = threadIdx.x;
    const int m0   = blockIdx.x * 128;
    const int n0   = blockIdx.y * 64;
    const int lane = tid & 31, w = tid >> 5;
    const int wm = w >> 1, wn = w & 1;
    const int mbase = wm * 32, nb = wn * 32;
    const int l16 = lane & 15, lhi = lane >> 4;

    v8f acc[2][2] = {};
    const int ksteps = K >> 5;
    for (int ks = 0; ks < ksteps; ++ks) {
        const int k0 = ks << 5;
        // --- stage A: 128 rows x 32 halves ---
        {
            const int r = tid >> 1, cb = (tid & 1) << 4;
            const _Float16* src = X + (size_t)(m0 + r) * K + k0 + cb;
            *(v8h*)&As[r*40 + cb]     = *(const v8h*)src;
            *(v8h*)&As[r*40 + cb + 8] = *(const v8h*)(src + 8);
            if (ks + 1 < ksteps)
                __builtin_prefetch(src + 32, 0, 1);   // global_prefetch_b8
        }
        // --- stage B transposed with f32->f16 convert: Bs[n][k] ---
        {
            const int n = tid & 63, kk = (tid >> 6) << 3;
            _Float16 tmp[8];
            #pragma unroll
            for (int k = 0; k < 8; ++k)
                tmp[k] = (_Float16)W[(size_t)(k0 + kk + k) * Nout + n0 + n];
            *(v8h*)&Bs[n*40 + kk] = *(v8h*)tmp;
        }
        __syncthreads();
        // --- fragments per ISA VGPR layout ---
        V16 a0, a1, b0, b1;
        const int kb  = lhi << 3;     // A: lanes<16 -> K {0..7,16..23}
        const int kbb = lhi << 3;     // B: lanes<16 -> K {0..7,16..23}
        const int ar0 = (mbase + l16) * 40,  ar1 = (mbase + 16 + l16) * 40;
        a0.h[0] = *(const v8h*)&As[ar0 + kb];      a0.h[1] = *(const v8h*)&As[ar0 + 16 + kb];
        a1.h[0] = *(const v8h*)&As[ar1 + kb];      a1.h[1] = *(const v8h*)&As[ar1 + 16 + kb];
        const int br0 = (nb + l16) * 40, br1 = (nb + 16 + l16) * 40;
        b0.h[0] = *(const v8h*)&Bs[br0 + kbb];     b0.h[1] = *(const v8h*)&Bs[br0 + 16 + kbb];
        b1.h[0] = *(const v8h*)&Bs[br1 + kbb];     b1.h[1] = *(const v8h*)&Bs[br1 + 16 + kbb];
        acc[0][0] = __builtin_amdgcn_wmma_f32_16x16x32_f16(false, a0.v, false, b0.v, (short)0, acc[0][0], false, false);
        acc[0][1] = __builtin_amdgcn_wmma_f32_16x16x32_f16(false, a0.v, false, b1.v, (short)0, acc[0][1], false, false);
        acc[1][0] = __builtin_amdgcn_wmma_f32_16x16x32_f16(false, a1.v, false, b0.v, (short)0, acc[1][0], false, false);
        acc[1][1] = __builtin_amdgcn_wmma_f32_16x16x32_f16(false, a1.v, false, b1.v, (short)0, acc[1][1], false, false);
        __syncthreads();
    }
    // --- epilogue: bias + relu + f16 store ---
    #pragma unroll
    for (int i = 0; i < 2; ++i) {
        #pragma unroll
        for (int j = 0; j < 2; ++j) {
            const int col = n0 + nb + j*16 + l16;
            const float bv = bias[col];
            #pragma unroll
            for (int r = 0; r < 8; ++r) {
                const int row = m0 + mbase + i*16 + (lhi << 3) + r;
                float vv = acc[i][j][r] + bv;
                if (relu) vv = fmaxf(vv, 0.f);
                Y[(size_t)row * Nout + col] = (_Float16)vv;
            }
        }
    }
}

// ===========================================================================
// psi last layer fused: y = act3 @ W(256x1024) + b, then
// u[b,c] += sum_m y[m,c] * h[b,m, c&31]     (LDS reduce, global atomics)
// ===========================================================================
__global__ __launch_bounds__(256) void k_psi_fused(
    const _Float16* __restrict__ act, const float* __restrict__ W,
    const float* __restrict__ bias, const float* __restrict__ h,
    float* __restrict__ u)
{
    __shared__ __align__(16) _Float16 As[128*40];
    __shared__ __align__(16) _Float16 Bs[64*40];
    __shared__ float hs[128*33];
    __shared__ float u_lds[1024];
    const int tid  = threadIdx.x;
    const int m0   = blockIdx.x * 128;
    const int bidx = m0 >> 12;            // 4096 points per batch element
    const int lane = tid & 31, w = tid >> 5;
    const int wm = w >> 1, wn = w & 1;
    const int mbase = wm * 32, nb = wn * 32;
    const int l16 = lane & 15, lhi = lane >> 4;

    for (int i = tid; i < 1024; i += 256) u_lds[i] = 0.f;
    for (int idx = tid; idx < 128*32; idx += 256) {
        const int r = idx >> 5, c = idx & 31;
        hs[r*33 + c] = h[(size_t)(m0 + r)*32 + c];
    }
    __syncthreads();

    for (int chunk = 0; chunk < 16; ++chunk) {
        const int col0 = chunk << 6;
        v8f acc[2][2] = {};
        for (int ks = 0; ks < 8; ++ks) {
            const int k0 = ks << 5;
            {   // stage A
                const int r = tid >> 1, cb = (tid & 1) << 4;
                const _Float16* src = act + (size_t)(m0 + r)*256 + k0 + cb;
                *(v8h*)&As[r*40 + cb]     = *(const v8h*)src;
                *(v8h*)&As[r*40 + cb + 8] = *(const v8h*)(src + 8);
            }
            {   // stage B^T with convert
                const int n = tid & 63, kk = (tid >> 6) << 3;
                _Float16 tmp[8];
                #pragma unroll
                for (int k = 0; k < 8; ++k)
                    tmp[k] = (_Float16)W[(size_t)(k0 + kk + k)*1024 + col0 + n];
                *(v8h*)&Bs[n*40 + kk] = *(v8h*)tmp;
            }
            __syncthreads();
            V16 a0, a1, b0, b1;
            const int kb = lhi << 3, kbb = lhi << 3;
            const int ar0 = (mbase + l16)*40, ar1 = (mbase + 16 + l16)*40;
            a0.h[0] = *(const v8h*)&As[ar0 + kb];  a0.h[1] = *(const v8h*)&As[ar0 + 16 + kb];
            a1.h[0] = *(const v8h*)&As[ar1 + kb];  a1.h[1] = *(const v8h*)&As[ar1 + 16 + kb];
            const int br0 = (nb + l16)*40, br1 = (nb + 16 + l16)*40;
            b0.h[0] = *(const v8h*)&Bs[br0 + kbb]; b0.h[1] = *(const v8h*)&Bs[br0 + 16 + kbb];
            b1.h[0] = *(const v8h*)&Bs[br1 + kbb]; b1.h[1] = *(const v8h*)&Bs[br1 + 16 + kbb];
            acc[0][0] = __builtin_amdgcn_wmma_f32_16x16x32_f16(false, a0.v, false, b0.v, (short)0, acc[0][0], false, false);
            acc[0][1] = __builtin_amdgcn_wmma_f32_16x16x32_f16(false, a0.v, false, b1.v, (short)0, acc[0][1], false, false);
            acc[1][0] = __builtin_amdgcn_wmma_f32_16x16x32_f16(false, a1.v, false, b0.v, (short)0, acc[1][0], false, false);
            acc[1][1] = __builtin_amdgcn_wmma_f32_16x16x32_f16(false, a1.v, false, b1.v, (short)0, acc[1][1], false, false);
            __syncthreads();
        }
        // contraction with h over the tile's 128 rows
        #pragma unroll
        for (int j = 0; j < 2; ++j) {
            const int c  = col0 + nb + j*16 + l16;
            const float bv = bias[c];
            const int ii = c & 31;
            float part = 0.f;
            #pragma unroll
            for (int i = 0; i < 2; ++i) {
                const int mloc = mbase + i*16 + (lhi << 3);
                #pragma unroll
                for (int r = 0; r < 8; ++r)
                    part += (acc[i][j][r] + bv) * hs[(mloc + r)*33 + ii];
            }
            atomicAdd(&u_lds[c], part);
        }
    }
    __syncthreads();
    for (int i = tid; i < 1024; i += 256)
        atomicAdd(&u[bidx*1024 + i], u_lds[i]);
}

// ===========================================================================
// phi last layer fused: y = act3 @ W + b, then
// v1[m,o] = sum_c y[m,c] * u[b,c] / N   (shfl_xor reduce; one writer per (m,o))
// ===========================================================================
__global__ __launch_bounds__(256) void k_phi_fused(
    const _Float16* __restrict__ act, const float* __restrict__ W,
    const float* __restrict__ bias, const float* __restrict__ u,
    float* __restrict__ v1)
{
    __shared__ __align__(16) _Float16 As[128*40];
    __shared__ __align__(16) _Float16 Bs[64*40];
    __shared__ float us[1024];
    __shared__ float v1s[128*33];
    const int tid  = threadIdx.x;
    const int m0   = blockIdx.x * 128;
    const int bidx = m0 >> 12;
    const int lane = tid & 31, w = tid >> 5;
    const int wm = w >> 1, wn = w & 1;
    const int mbase = wm * 32, nb = wn * 32;
    const int l16 = lane & 15, lhi = lane >> 4;

    for (int i = tid; i < 1024; i += 256)
        us[i] = u[bidx*1024 + i] * (1.f / (float)NPTS);
    __syncthreads();

    for (int chunk = 0; chunk < 16; ++chunk) {
        const int col0 = chunk << 6;
        v8f acc[2][2] = {};
        for (int ks = 0; ks < 8; ++ks) {
            const int k0 = ks << 5;
            {
                const int r = tid >> 1, cb = (tid & 1) << 4;
                const _Float16* src = act + (size_t)(m0 + r)*256 + k0 + cb;
                *(v8h*)&As[r*40 + cb]     = *(const v8h*)src;
                *(v8h*)&As[r*40 + cb + 8] = *(const v8h*)(src + 8);
            }
            {
                const int n = tid & 63, kk = (tid >> 6) << 3;
                _Float16 tmp[8];
                #pragma unroll
                for (int k = 0; k < 8; ++k)
                    tmp[k] = (_Float16)W[(size_t)(k0 + kk + k)*1024 + col0 + n];
                *(v8h*)&Bs[n*40 + kk] = *(v8h*)tmp;
            }
            __syncthreads();
            V16 a0, a1, b0, b1;
            const int kb = lhi << 3, kbb = lhi << 3;
            const int ar0 = (mbase + l16)*40, ar1 = (mbase + 16 + l16)*40;
            a0.h[0] = *(const v8h*)&As[ar0 + kb];  a0.h[1] = *(const v8h*)&As[ar0 + 16 + kb];
            a1.h[0] = *(const v8h*)&As[ar1 + kb];  a1.h[1] = *(const v8h*)&As[ar1 + 16 + kb];
            const int br0 = (nb + l16)*40, br1 = (nb + 16 + l16)*40;
            b0.h[0] = *(const v8h*)&Bs[br0 + kbb]; b0.h[1] = *(const v8h*)&Bs[br0 + 16 + kbb];
            b1.h[0] = *(const v8h*)&Bs[br1 + kbb]; b1.h[1] = *(const v8h*)&Bs[br1 + 16 + kbb];
            acc[0][0] = __builtin_amdgcn_wmma_f32_16x16x32_f16(false, a0.v, false, b0.v, (short)0, acc[0][0], false, false);
            acc[0][1] = __builtin_amdgcn_wmma_f32_16x16x32_f16(false, a0.v, false, b1.v, (short)0, acc[0][1], false, false);
            acc[1][0] = __builtin_amdgcn_wmma_f32_16x16x32_f16(false, a1.v, false, b0.v, (short)0, acc[1][0], false, false);
            acc[1][1] = __builtin_amdgcn_wmma_f32_16x16x32_f16(false, a1.v, false, b1.v, (short)0, acc[1][1], false, false);
            __syncthreads();
        }
        // whole wave sub-tile (32 cols) maps to a single o = chunk*2 + wn
        const int o  = (col0 + nb) >> 5;
        const int c0 = col0 + nb + l16, c1 = c0 + 16;
        const float u0 = us[c0], u1 = us[c1];
        const float b0v = bias[c0], b1v = bias[c1];
        #pragma unroll
        for (int i = 0; i < 2; ++i) {
            #pragma unroll
            for (int r = 0; r < 8; ++r) {
                float vacc = (acc[i][0][r] + b0v) * u0 + (acc[i][1][r] + b1v) * u1;
                vacc += __shfl_xor(vacc, 1);
                vacc += __shfl_xor(vacc, 2);
                vacc += __shfl_xor(vacc, 4);
                vacc += __shfl_xor(vacc, 8);   // lanes 0 and 16 hold the two row sums
                if (l16 == 0)
                    v1s[(mbase + i*16 + (lhi << 3) + r)*33 + o] = vacc;
            }
        }
    }
    __syncthreads();
    for (int idx = tid; idx < 128*32; idx += 256) {
        const int r = idx >> 5, c = idx & 31;
        v1[(size_t)(m0 + r)*32 + c] = v1s[r*33 + c];
    }
}

// ===========================================================================
// s = v1 + h @ w_l + b_l ; accumulate per-channel sum / sumsq for batchnorm
// ===========================================================================
__global__ __launch_bounds__(256) void k_v2_stats(
    const float* __restrict__ h32, const float* __restrict__ Wl,
    const float* __restrict__ bl, float* __restrict__ s,
    float* __restrict__ stats)
{
    __shared__ float Ws[1024], bs[32], ssum[32], ssq[32];
    const int tid = threadIdx.x;
    for (int i = tid; i < 1024; i += 256) Ws[i] = Wl[i];
    if (tid < 32) { bs[tid] = bl[tid]; ssum[tid] = 0.f; ssq[tid] = 0.f; }
    __syncthreads();
    const int p = blockIdx.x * 256 + tid;
    float x[32];
    #pragma unroll
    for (int i = 0; i < 32; ++i) x[i] = h32[(size_t)p*32 + i];
    #pragma unroll
    for (int o = 0; o < 32; ++o) {
        float v2 = bs[o];
        #pragma unroll
        for (int i = 0; i < 32; ++i) v2 += x[i] * Ws[i*32 + o];
        const float sv = s[(size_t)p*32 + o] + v2;
        s[(size_t)p*32 + o] = sv;
        atomicAdd(&ssum[o], sv);
        atomicAdd(&ssq[o],  sv * sv);
    }
    __syncthreads();
    if (tid < 32) {
        atomicAdd(&stats[tid],      ssum[tid]);
        atomicAdd(&stats[32 + tid], ssq[tid]);
    }
}

// ===========================================================================
// batchnorm apply (+ optional relu) -> new h (f32 + f16)
// ===========================================================================
__global__ __launch_bounds__(256) void k_bn_apply(
    const float* __restrict__ s, const float* __restrict__ stats,
    const float* __restrict__ gamma, const float* __restrict__ beta,
    float* __restrict__ h32, _Float16* __restrict__ h16, int relu)
{
    __shared__ float mu_s[32], sc_s[32], bt_s[32];
    const int tid = threadIdx.x;
    if (tid < 32) {
        const float cnt = (float)MTOT;
        const float mu  = stats[tid] / cnt;
        const float var = stats[32 + tid] / cnt - mu * mu;
        mu_s[tid] = mu;
        sc_s[tid] = rsqrtf(var + BN_EPS) * gamma[tid];
        bt_s[tid] = beta[tid];
    }
    __syncthreads();
    const int p = blockIdx.x * 256 + tid;
    #pragma unroll
    for (int o = 0; o < 32; ++o) {
        float y = (s[(size_t)p*32 + o] - mu_s[o]) * sc_s[o] + bt_s[o];
        if (relu) y = fmaxf(y, 0.f);
        h32[(size_t)p*32 + o] = y;
        h16[(size_t)p*32 + o] = (_Float16)y;
    }
}

// ===========================================================================
// head: out = relu(h @ W1(32x128) + b1) @ W2(128x1) + b2
// ===========================================================================
__global__ __launch_bounds__(256) void k_head(
    const float* __restrict__ h32, const float* __restrict__ W1,
    const float* __restrict__ b1, const float* __restrict__ W2,
    const float* __restrict__ b2, float* __restrict__ out)
{
    __shared__ float W1s[4096], b1s[128], W2s[128];
    const int tid = threadIdx.x;
    for (int i = tid; i < 4096; i += 256) W1s[i] = W1[i];
    if (tid < 128) { b1s[tid] = b1[tid]; W2s[tid] = W2[tid]; }
    __syncthreads();
    const int p = blockIdx.x * 256 + tid;
    float x[32];
    #pragma unroll
    for (int i = 0; i < 32; ++i) x[i] = h32[(size_t)p*32 + i];
    float acc = b2[0];
    for (int j = 0; j < 128; ++j) {
        float t = b1s[j];
        #pragma unroll
        for (int i = 0; i < 32; ++i) t += x[i] * W1s[i*128 + j];
        acc += fmaxf(t, 0.f) * W2s[j];
    }
    out[p] = acc;
}

__global__ void k_zero(float* p, int n)
{
    const int i = blockIdx.x * 256 + threadIdx.x;
    if (i < n) p[i] = 0.f;
}

// ===========================================================================
// host side
// ===========================================================================
extern "C" void kernel_launch(void* const* d_in, const int* in_sizes, int n_in,
                              void* d_out, int out_size, void* d_ws, size_t ws_size,
                              hipStream_t stream)
{
    (void)in_sizes; (void)n_in; (void)out_size; (void)ws_size;
    // JAX pytree flatten order (sorted dict keys):
    // params{ bn0..bn3, fc0, fc1, fc2, phi0..phi3, psi0..psi3, w0..w3 }, v
    #define IN_F(i) ((const float*)d_in[(i)])
    const float* bn_g[4] = { IN_F(0), IN_F(2), IN_F(4), IN_F(6) };
    const float* bn_b[4] = { IN_F(1), IN_F(3), IN_F(5), IN_F(7) };
    const float* fc0W = IN_F(8),  *fc0b = IN_F(9);
    const float* fc1W = IN_F(10), *fc1b = IN_F(11);
    const float* fc2W = IN_F(12), *fc2b = IN_F(13);
    const int PHI0 = 14, PSI0 = 46, WL0 = 78, VIN = 86;
    const float* v = IN_F(VIN);
    float* out = (float*)d_out;

    // workspace layout
    char* wsp = (char*)d_ws;
    float*    h32  = (float*)wsp;               wsp += (size_t)MTOT*32*4;
    _Float16* h16  = (_Float16*)wsp;            wsp += (size_t)MTOT*32*2;
    _Float16* act1 = (_Float16*)wsp;            wsp += (size_t)MTOT*64*2;
    _Float16* act2 = (_Float16*)wsp;            wsp += (size_t)MTOT*128*2;
    _Float16* act3 = (_Float16*)wsp;            wsp += (size_t)MTOT*256*2;
    float*    v1s  = (float*)wsp;               wsp += (size_t)MTOT*32*4;
    float*    u    = (float*)wsp;               wsp += (size_t)8*1024*4;
    float*    stats= (float*)wsp;               wsp += 64*4;

    const dim3 blk(256);
    const dim3 gPts(MTOT/256);        // 128 blocks, one point per thread
    const dim3 gTile(MTOT/128);       // 256 GEMM tiles

    k_fc0<<<gPts, blk, 0, stream>>>(v, fc0W, fc0b, h32, h16);

    for (int l = 0; l < 4; ++l) {
        const int phi = PHI0 + 8*l, psi = PSI0 + 8*l;
        k_zero<<<(8192+64+255)/256, blk, 0, stream>>>(u, 8192 + 64);

        // ---- psi branch -> u ----
        k_mlp_first<<<gPts, blk, 0, stream>>>(v, IN_F(psi+0), IN_F(psi+1), act1);
        k_gemm_wmma<<<dim3(MTOT/128, 2), blk, 0, stream>>>(act1, IN_F(psi+2), IN_F(psi+3), act2,  64, 128, 1);
        k_gemm_wmma<<<dim3(MTOT/128, 4), blk, 0, stream>>>(act2, IN_F(psi+4), IN_F(psi+5), act3, 128, 256, 1);
        k_psi_fused<<<gTile, blk, 0, stream>>>(act3, IN_F(psi+6), IN_F(psi+7), h32, u);

        // ---- phi branch -> v1 (already / N) ----
        k_mlp_first<<<gPts, blk, 0, stream>>>(v, IN_F(phi+0), IN_F(phi+1), act1);
        k_gemm_wmma<<<dim3(MTOT/128, 2), blk, 0, stream>>>(act1, IN_F(phi+2), IN_F(phi+3), act2,  64, 128, 1);
        k_gemm_wmma<<<dim3(MTOT/128, 4), blk, 0, stream>>>(act2, IN_F(phi+4), IN_F(phi+5), act3, 128, 256, 1);
        k_phi_fused<<<gTile, blk, 0, stream>>>(act3, IN_F(phi+6), IN_F(phi+7), u, v1s);

        // ---- s = v1 + h@w_l + b_l, batchnorm, relu (l<3) ----
        k_v2_stats<<<gPts, blk, 0, stream>>>(h32, IN_F(WL0+2*l), IN_F(WL0+2*l+1), v1s, stats);
        k_bn_apply<<<gPts, blk, 0, stream>>>(v1s, stats, bn_g[l], bn_b[l], h32, h16, (l < 3) ? 1 : 0);
    }

    k_head<<<gPts, blk, 0, stream>>>(h32, fc1W, fc1b, fc2W, fc2b, out);
    #undef IN_F
}
